// ResNet_8572754722937
// MI455X (gfx1250) — matrix operand order repaired and checked by reading, not compiled, not run on hardware
//
#include <hip/hip_runtime.h>
#include <hip/hip_bf16.h>

// ---------------------------------------------------------------------------
// MI455X (gfx1250) implementation of the sparse/dense ResNet reference.
//   * activations: bf16 NHWC (channel-contiguous -> b128 loads of 8 ch/tap)
//   * convs: implicit GEMM on v_wmma_f32_16x16x32_bf16
//       M = output pixels, N = Cout, K = Cin*kh*kw (tap-major, cin-minor)
//       one wave: MT=2 M-tiles x NT=4 N-tiles (8 accumulators); every B
//       (weight) fragment is reused across both M-tiles -> 2x WMMA/byte
//       all spatial sizes are powers of two -> shift/mask addressing, no divs
//   * batch-norm (training mode): contiguous float4 read + LDS-atomic channel
//     bins -> per-channel scale/shift -> fused apply(+res,+relu) emitting bf16
//   * head: 1x1 conv + global avgpool + softmax fused
// ---------------------------------------------------------------------------

typedef __attribute__((ext_vector_type(16))) __bf16 v16bf;
typedef __attribute__((ext_vector_type(8)))  __bf16 v8bf;
typedef __attribute__((ext_vector_type(4)))  __bf16 v4bf;
typedef __attribute__((ext_vector_type(8)))  float  v8f;

static inline int cdiv(int a, int b) { return (a + b - 1) / b; }
static inline long long llmin(long long a, long long b) { return a < b ? a : b; }

// ---------------- weight pack: OIHW f32 -> [Cout][(ky*KW+kx)*Cin + cin] bf16
__global__ __launch_bounds__(256) void pack_weights(
    const float* __restrict__ w, __bf16* __restrict__ wpk,
    int Cout, int Cin, int KH, int KW)
{
  const int Ktot = KH * KW * Cin;
  const int total = Cout * Ktot;
  int idx = blockIdx.x * blockDim.x + threadIdx.x;
  if (idx >= total) return;
  int k  = idx % Ktot;
  int co = idx / Ktot;
  int cin = k % Cin;
  int tap = k / Cin;
  int ky = tap / KW, kx = tap % KW;
  wpk[idx] = (__bf16)w[((size_t)(co * Cin + cin) * KH + ky) * KW + kx];
}

// ---------------- implicit-GEMM conv
// one wave: MT*16 pixels x NT*16 couts; square pow2 spatial sizes.
template <int MT, int NT>
__global__ __launch_bounds__(128) void conv_wmma(
    const __bf16* __restrict__ in,   // NHWC bf16, H=W=1<<lsin
    const __bf16* __restrict__ wpk,  // [Cout][Ktot] bf16
    float* __restrict__ out,         // NHWC f32, H=W=1<<lsout
    int N, int lsin, int Cin, int Cout, int lsout,
    int KH, int KW, int stride, int pad)
{
  const int Hin  = 1 << lsin;
  const int Hout = 1 << lsout;
  const int M = N << (2 * lsout);
  const int mtiles = (M + 15) >> 4;
  const int lane = threadIdx.x & 31;
  const int wave = threadIdx.x >> 5;
  const int mtile0 = (blockIdx.x * 4 + wave) * MT;  // wave-uniform early-out
  if (mtile0 >= mtiles) return;
  const int ctile0 = blockIdx.y * NT;
  const int ntiles = Cout >> 4;

  const int Ktot = KH * KW * Cin;
  const int row = lane & 15;
  const int hi  = lane >> 4;
  const int kbaseA = hi ? 8 : 0;    // A frag: lanes<16 hold k {0..7,16..23}
  const int kbaseB = hi ? 8 : 0;    // B frag: lanes<16 hold k {0..7,16..23}

  // per-M-tile pixel state; invalid rows alias pixel 0 (stores are guarded)
  int iy0[MT], ix0[MT];
  const __bf16* inb[MT];
  #pragma unroll
  for (int mt = 0; mt < MT; ++mt) {
    const int m = ((mtile0 + mt) << 4) + row;
    const int ms = (m < M) ? m : 0;
    const int ox = ms & (Hout - 1);
    const int oy = (ms >> lsout) & (Hout - 1);
    const int n  = ms >> (2 * lsout);
    iy0[mt] = oy * stride - pad;
    ix0[mt] = ox * stride - pad;
    inb[mt] = in + (((size_t)n << (2 * lsin)) * Cin);
  }

  const __bf16* wrow[NT];
  #pragma unroll
  for (int j = 0; j < NT; ++j)
    wrow[j] = wpk + (size_t)((((ctile0 + j) << 4) + row)) * Ktot;

  v8f acc[MT][NT] = {};

  auto loadA8 = [&](int mt, int cin_, int ky_, int kx_) -> v8bf {
    v8bf r = {};
    const int iy = iy0[mt] + ky_, ix = ix0[mt] + kx_;
    if ((unsigned)iy < (unsigned)Hin && (unsigned)ix < (unsigned)Hin)
      r = *(const v8bf*)(inb[mt] + (((size_t)iy << lsin) + ix) * Cin + cin_);
    return r;
  };

  if ((Cin & 7) == 0) {
    // -------- fast path: contiguous 8-channel runs, incremental tap/cin ----
    int cinA = kbaseA, kyA = 0, kxA = 0;    // state for k = k0 + kbaseA
    for (int k0 = 0; k0 < Ktot; k0 += 32) {
      int c2 = cinA + 16, ky2 = kyA, kx2 = kxA;
      while (c2 >= Cin) { c2 -= Cin; if (++kx2 == KW) { kx2 = 0; ++ky2; } }
      const bool h0 = (k0 + kbaseA) < Ktot;
      const bool h1 = (k0 + kbaseA + 16) < Ktot;

      v16bf a[MT] = {};
      #pragma unroll
      for (int mt = 0; mt < MT; ++mt) {
        if (h0) {
          const v8bf v = loadA8(mt, cinA, kyA, kxA);
          #pragma unroll
          for (int i = 0; i < 8; ++i) a[mt][i] = v[i];
        }
        if (h1) {
          const v8bf v = loadA8(mt, c2, ky2, kx2);
          #pragma unroll
          for (int i = 0; i < 8; ++i) a[mt][8 + i] = v[i];
        }
      }

      const int kb = k0 + kbaseB;
      if (kb + 47 < Ktot) __builtin_prefetch(wrow[0] + kb + 32, 0, 0);
      #pragma unroll
      for (int j = 0; j < NT; ++j) {
        if (ctile0 + j < ntiles) {
          v16bf bfr = {};
          if (kb + 23 < Ktot) {
            const v8bf w0 = *(const v8bf*)(wrow[j] + kb);
            const v8bf w1 = *(const v8bf*)(wrow[j] + kb + 16);
            #pragma unroll
            for (int i = 0; i < 8; ++i) { bfr[i] = w0[i]; bfr[8 + i] = w1[i]; }
          } else {
            #pragma unroll
            for (int i = 0; i < 16; ++i) {
              const int kk = kb + (i < 8 ? i : 8 + i);
              if (kk < Ktot) bfr[i] = wrow[j][kk];
            }
          }
          #pragma unroll
          for (int mt = 0; mt < MT; ++mt)
            acc[mt][j] = __builtin_amdgcn_wmma_f32_16x16x32_bf16(
                false, a[mt], false, bfr, (short)0, acc[mt][j], false, false);
        }
      }
      cinA += 32;
      while (cinA >= Cin) { cinA -= Cin; if (++kxA == KW) { kxA = 0; ++kyA; } }
    }
  } else {
    // -------- generic path (init conv, Cin == 1): tap == k ------------------
    for (int k0 = 0; k0 < Ktot; k0 += 32) {
      v16bf a[MT] = {};
      #pragma unroll
      for (int i = 0; i < 16; ++i) {
        const int kk = k0 + kbaseA + (i < 8 ? i : 8 + i);
        if (kk < Ktot) {
          const int ky = kk / KW, kx = kk - ky * KW;
          #pragma unroll
          for (int mt = 0; mt < MT; ++mt) {
            const int iy = iy0[mt] + ky, ix = ix0[mt] + kx;
            if ((unsigned)iy < (unsigned)Hin && (unsigned)ix < (unsigned)Hin)
              a[mt][i] = inb[mt][(((size_t)iy << lsin) + ix)];
          }
        }
      }
      const int kb = k0 + kbaseB;
      #pragma unroll
      for (int j = 0; j < NT; ++j) {
        if (ctile0 + j < ntiles) {
          v16bf bfr = {};
          #pragma unroll
          for (int i = 0; i < 16; ++i) {
            const int kk = kb + (i < 8 ? i : 8 + i);
            if (kk < Ktot) bfr[i] = wrow[j][kk];
          }
          #pragma unroll
          for (int mt = 0; mt < MT; ++mt)
            acc[mt][j] = __builtin_amdgcn_wmma_f32_16x16x32_bf16(
                false, a[mt], false, bfr, (short)0, acc[mt][j], false, false);
        }
      }
    }
  }

  // C/D layout: VGPR r -> pixel (mtile*16 + hi*8 + r); column = cout
  #pragma unroll
  for (int mt = 0; mt < MT; ++mt) {
    #pragma unroll
    for (int j = 0; j < NT; ++j) {
      if (ctile0 + j < ntiles) {
        const int cout = ((ctile0 + j) << 4) + row;
        #pragma unroll
        for (int r = 0; r < 8; ++r) {
          const int mm = ((mtile0 + mt) << 4) + (hi << 3) + r;
          if (mm < M) {
            const int oxx = mm & (Hout - 1);
            const int oyy = (mm >> lsout) & (Hout - 1);
            const int nn  = mm >> (2 * lsout);
            out[((((size_t)nn << lsout) + oyy) * (size_t)Hout + oxx) * Cout + cout] =
                acc[mt][j][r];
          }
        }
      }
    }
  }
}

// ---------------- elementwise / reduction kernels
__global__ __launch_bounds__(256) void cvt_f32_bf16(
    const float* __restrict__ in, __bf16* __restrict__ out, long long n)
{
  long long i = ((long long)blockIdx.x * blockDim.x + threadIdx.x) * 4;
  if (i >= n) return;
  const float4 v = *(const float4*)(in + i);
  v4bf o;
  o[0] = (__bf16)v.x; o[1] = (__bf16)v.y; o[2] = (__bf16)v.z; o[3] = (__bf16)v.w;
  *(v4bf*)(out + i) = o;
}

__global__ __launch_bounds__(256) void zero_f32(float* __restrict__ p, int n)
{
  int i = blockIdx.x * blockDim.x + threadIdx.x;
  if (i < n) p[i] = 0.f;
}

// contiguous float4 stream -> LDS channel bins (ds_add_f32) -> global atomics
__global__ __launch_bounds__(256) void bn_partial(
    const float* __restrict__ x, float* __restrict__ sums, long long total, int C)
{
  __shared__ float bs[256], bs2[256];
  bs[threadIdx.x] = 0.f;
  bs2[threadIdx.x] = 0.f;
  __syncthreads();
  const long long stride = (long long)gridDim.x * blockDim.x * 4;
  long long i = ((long long)blockIdx.x * blockDim.x + threadIdx.x) * 4;
  int c = (int)(i % C);                    // C % 4 == 0 -> c stays 4-aligned
  const int step = (int)(stride % C);
  for (; i < total; i += stride) {
    const float4 v = *(const float4*)(x + i);
    atomicAdd(&bs[c + 0], v.x); atomicAdd(&bs2[c + 0], v.x * v.x);
    atomicAdd(&bs[c + 1], v.y); atomicAdd(&bs2[c + 1], v.y * v.y);
    atomicAdd(&bs[c + 2], v.z); atomicAdd(&bs2[c + 2], v.z * v.z);
    atomicAdd(&bs[c + 3], v.w); atomicAdd(&bs2[c + 3], v.w * v.w);
    c += step;
    if (c >= C) c -= C;
  }
  __syncthreads();
  if (threadIdx.x < (unsigned)C) {
    atomicAdd(&sums[threadIdx.x],     bs[threadIdx.x]);
    atomicAdd(&sums[C + threadIdx.x], bs2[threadIdx.x]);
  }
}

// fold mean/var + gamma/beta into per-channel scale/shift
__global__ __launch_bounds__(128) void bn_finalize(
    const float* __restrict__ sums, const float* __restrict__ g,
    const float* __restrict__ b, float* __restrict__ scale,
    float* __restrict__ shift, long long NHW, int C)
{
  int c = blockIdx.x * blockDim.x + threadIdx.x;
  if (c < C) {
    float m  = sums[c] / (float)NHW;
    float v  = sums[C + c] / (float)NHW - m * m;
    float rs = rsqrtf(fmaxf(v, 0.f) + 1e-5f);
    float sc = rs * g[c];
    scale[c] = sc;
    shift[c] = b[c] - m * sc;
  }
}

// y = x*scale + shift (+res) (relu?) -> bf16, 4-wide
__global__ __launch_bounds__(256) void bn_apply(
    const float* __restrict__ x, const float* __restrict__ scale,
    const float* __restrict__ shift, const __bf16* __restrict__ res,
    __bf16* __restrict__ out, long long total, int C, int relu)
{
  long long i = ((long long)blockIdx.x * blockDim.x + threadIdx.x) * 4;
  if (i >= total) return;
  int c = (int)(i % C);
  const float4 v = *(const float4*)(x + i);
  float o0 = v.x * scale[c + 0] + shift[c + 0];
  float o1 = v.y * scale[c + 1] + shift[c + 1];
  float o2 = v.z * scale[c + 2] + shift[c + 2];
  float o3 = v.w * scale[c + 3] + shift[c + 3];
  if (res) {
    const v4bf r = *(const v4bf*)(res + i);
    o0 += (float)r[0]; o1 += (float)r[1]; o2 += (float)r[2]; o3 += (float)r[3];
  }
  if (relu) {
    o0 = fmaxf(o0, 0.f); o1 = fmaxf(o1, 0.f);
    o2 = fmaxf(o2, 0.f); o3 = fmaxf(o3, 0.f);
  }
  v4bf o;
  o[0] = (__bf16)o0; o[1] = (__bf16)o1; o[2] = (__bf16)o2; o[3] = (__bf16)o3;
  *(v4bf*)(out + i) = o;
}

// concat planes: [(p*B+b),H,W,C] -> [b,H,W,P*C], 8 bf16 per thread
__global__ __launch_bounds__(256) void merge_planes(
    const __bf16* __restrict__ h, __bf16* __restrict__ out,
    int P, int B, int H, int W, int C)
{
  const long long total = (long long)P * B * H * W * C;
  long long k = ((long long)blockIdx.x * blockDim.x + threadIdx.x) * 8;
  if (k >= total) return;
  int c = (int)(k % C);
  long long t = k / C;
  int x = (int)(t % W); t /= W;
  int y = (int)(t % H); t /= H;
  int b = (int)(t % B);
  int p = (int)(t / B);
  const v8bf v = *(const v8bf*)(h + k);
  *(v8bf*)(out + ((size_t)(b * H + y) * W + x) * (size_t)(P * C) + p * C + c) = v;
}

// 1x1 conv (C->3) + global avgpool + softmax, one block per image
__global__ __launch_bounds__(256) void head_kernel(
    const __bf16* __restrict__ act, const float* __restrict__ w,
    float* __restrict__ out, int HW, int C)
{
  const int b = blockIdx.x;
  const __bf16* ab = act + (size_t)b * HW * C;
  float a0 = 0.f, a1 = 0.f, a2 = 0.f;
  for (int i = threadIdx.x; i < HW * C; i += blockDim.x) {
    int c = i % C;
    float v = (float)ab[i];
    a0 += v * w[c];
    a1 += v * w[C + c];
    a2 += v * w[2 * C + c];
  }
  __shared__ float sh[3][256];
  sh[0][threadIdx.x] = a0; sh[1][threadIdx.x] = a1; sh[2][threadIdx.x] = a2;
  __syncthreads();
  for (int o = 128; o > 0; o >>= 1) {
    if (threadIdx.x < o) {
      sh[0][threadIdx.x] += sh[0][threadIdx.x + o];
      sh[1][threadIdx.x] += sh[1][threadIdx.x + o];
      sh[2][threadIdx.x] += sh[2][threadIdx.x + o];
    }
    __syncthreads();
  }
  if (threadIdx.x == 0) {
    float l0 = sh[0][0] / HW, l1 = sh[1][0] / HW, l2 = sh[2][0] / HW;
    float mx = fmaxf(l0, fmaxf(l1, l2));
    float e0 = expf(l0 - mx), e1 = expf(l1 - mx), e2 = expf(l2 - mx);
    float inv = 1.f / (e0 + e1 + e2);
    out[b * 3 + 0] = e0 * inv;
    out[b * 3 + 1] = e1 * inv;
    out[b * 3 + 2] = e2 * inv;
  }
}

// ---------------------------------------------------------------------------
// Host orchestration
// ---------------------------------------------------------------------------
namespace {

struct Ctx {
  hipStream_t s;
  float*  F32;       // f32 conv scratch (max 12*512*512*16 f32)
  __bf16* B[4];      // bf16 activation pools
  __bf16* WPK;       // packed weights
  float*  STAT;      // [0..2C) sums, +512 scale, +768 shift
};

void do_conv(const Ctx& c, const __bf16* in, const float* w, float* f32out,
             int N, int lsin, int Cin, int Cout,
             int KH, int KW, int stride, int pad, int lsout)
{
  const int Ktot = KH * KW * Cin;
  pack_weights<<<cdiv(Cout * Ktot, 256), 256, 0, c.s>>>(w, c.WPK, Cout, Cin, KH, KW);
  const int M = N << (2 * lsout);
  dim3 grid(cdiv(cdiv(M, 16), 8), cdiv(Cout / 16, 4));   // MT=2, 4 waves/block
  conv_wmma<2, 4><<<grid, 128, 0, c.s>>>(in, c.WPK, f32out,
                                         N, lsin, Cin, Cout, lsout,
                                         KH, KW, stride, pad);
}

void do_bn(const Ctx& c, const float* x, const float* g, const float* b,
           const __bf16* res, __bf16* out, long long NHW, int C, int relu)
{
  const long long total = NHW * C;
  zero_f32<<<cdiv(2 * C, 256), 256, 0, c.s>>>(c.STAT, 2 * C);
  const int nb = (int)llmin(2048, (total / 4 + 255) / 256);
  bn_partial<<<nb, 256, 0, c.s>>>(x, c.STAT, total, C);
  bn_finalize<<<cdiv(C, 128), 128, 0, c.s>>>(c.STAT, g, b,
                                             c.STAT + 512, c.STAT + 768, NHW, C);
  bn_apply<<<(int)((total / 4 + 255) / 256), 256, 0, c.s>>>(
      x, c.STAT + 512, c.STAT + 768, res, out, total, C, relu);
}

} // namespace

extern "C" void kernel_launch(void* const* d_in, const int* in_sizes, int n_in,
                              void* d_out, int out_size, void* d_ws, size_t ws_size,
                              hipStream_t stream)
{
  if (n_in < 74) return;   // x + 73 flattened params (insertion order, Nones skipped)

  const size_t F32B = (size_t)12 * 512 * 512 * 16 * sizeof(float);
  const size_t PBB  = (size_t)12 * 512 * 512 * 16 * sizeof(__bf16);
  const size_t NEED = F32B + 4 * PBB + (2u << 20) + 4096;
  if (ws_size < NEED) return;

  char* ws = (char*)d_ws;
  Ctx c;
  c.s   = stream;
  c.F32 = (float*)ws;
  size_t off = F32B;
  for (int i = 0; i < 4; ++i) { c.B[i] = (__bf16*)(ws + off); off += PBB; }
  c.WPK  = (__bf16*)(ws + off); off += (2u << 20);
  c.STAT = (float*)(ws + off);

  auto W_ = [&](int i) { return (const float*)d_in[i]; };

  // ---- pre-merge, N = P*B = 12 (weights shared across planes) ----
  cvt_f32_bf16<<<cdiv(3145728 / 4, 256), 256, 0, stream>>>(W_(0), c.B[0], 3145728LL);
  do_conv(c, c.B[0], W_(1), c.F32, 12, 9, 1, 16, 5, 5, 1, 2, 9);     // init 5x5, no BN
  cvt_f32_bf16<<<cdiv(50331648 / 4, 256), 256, 0, stream>>>(c.F32, c.B[1], 50331648LL);

  // sparse stride-1: conv2x2(pad lo 0) -> BN+ReLU -> conv2x2 -> BN -> +res
  auto sblock1 = [&](int iw, const __bf16* xin, __bf16* tmp, __bf16* outb,
                     int N, int ls, int C) {
    const long long NHW = (long long)N << (2 * ls);
    do_conv(c, xin, W_(iw + 0), c.F32, N, ls, C, C, 2, 2, 1, 0, ls);
    do_bn(c, c.F32, W_(iw + 1), W_(iw + 2), nullptr, tmp, NHW, C, 1);
    do_conv(c, tmp, W_(iw + 3), c.F32, N, ls, C, C, 2, 2, 1, 0, ls);
    do_bn(c, c.F32, W_(iw + 4), W_(iw + 5), xin, outb, NHW, C, 0);
  };
  // sparse stride-2: residual = conv2x2 s2 (no BN)
  auto sblock2 = [&](int iw, const __bf16* xin, __bf16* resb, __bf16* tmp,
                     __bf16* outb, int N, int ls, int Cin, int Cout) {
    const int lso = ls - 1;
    const long long NHWo = (long long)N << (2 * lso);
    do_conv(c, xin, W_(iw + 6), c.F32, N, ls, Cin, Cout, 2, 2, 2, 0, lso);
    cvt_f32_bf16<<<(int)((NHWo * Cout / 4 + 255) / 256), 256, 0, stream>>>(
        c.F32, resb, NHWo * Cout);
    do_conv(c, xin, W_(iw + 0), c.F32, N, ls, Cin, Cout, 2, 2, 2, 0, lso);
    do_bn(c, c.F32, W_(iw + 1), W_(iw + 2), nullptr, tmp, NHWo, Cout, 1);
    do_conv(c, tmp, W_(iw + 3), c.F32, N, lso, Cout, Cout, 2, 2, 1, 0, lso);
    do_bn(c, c.F32, W_(iw + 4), W_(iw + 5), resb, outb, NHWo, Cout, 0);
  };

  sblock1(2,  c.B[1], c.B[2], c.B[3], 12, 9, 16);
  sblock2(8,  c.B[3], c.B[1], c.B[0], c.B[2], 12, 9, 16, 32);
  sblock1(15, c.B[2], c.B[0], c.B[3], 12, 8, 32);
  sblock2(21, c.B[3], c.B[1], c.B[0], c.B[2], 12, 8, 32, 48);

  // concat planes -> [4,128,128,144]
  merge_planes<<<cdiv(12 * 128 * 128 * 48 / 8, 256), 256, 0, stream>>>(
      c.B[2], c.B[0], 3, 4, 128, 128, 48);

  // dense stride-1: conv3x3 -> BN+ReLU -> conv3x3 -> BN -> relu(+res)
  auto dblock1 = [&](int iw, const __bf16* xin, __bf16* tmp, __bf16* outb,
                     int N, int ls, int C) {
    const long long NHW = (long long)N << (2 * ls);
    do_conv(c, xin, W_(iw + 0), c.F32, N, ls, C, C, 3, 3, 1, 1, ls);
    do_bn(c, c.F32, W_(iw + 1), W_(iw + 2), nullptr, tmp, NHW, C, 1);
    do_conv(c, tmp, W_(iw + 3), c.F32, N, ls, C, C, 3, 3, 1, 1, ls);
    do_bn(c, c.F32, W_(iw + 4), W_(iw + 5), xin, outb, NHW, C, 1);
  };
  // dense stride-2: residual = BN(conv1x1 s2)
  auto dblock2 = [&](int iw, const __bf16* xin, __bf16* resb, __bf16* tmp,
                     __bf16* outb, int N, int ls, int Cin, int Cout) {
    const int lso = ls - 1;
    const long long NHWo = (long long)N << (2 * lso);
    do_conv(c, xin, W_(iw + 6), c.F32, N, ls, Cin, Cout, 1, 1, 2, 0, lso);
    do_bn(c, c.F32, W_(iw + 7), W_(iw + 8), nullptr, resb, NHWo, Cout, 0);
    do_conv(c, xin, W_(iw + 0), c.F32, N, ls, Cin, Cout, 3, 3, 2, 1, lso);
    do_bn(c, c.F32, W_(iw + 1), W_(iw + 2), nullptr, tmp, NHWo, Cout, 1);
    do_conv(c, tmp, W_(iw + 3), c.F32, N, lso, Cout, Cout, 3, 3, 1, 1, lso);
    do_bn(c, c.F32, W_(iw + 4), W_(iw + 5), resb, outb, NHWo, Cout, 1);
  };

  dblock1(28, c.B[0], c.B[1], c.B[2], 4, 7, 144);
  dblock2(34, c.B[2], c.B[1], c.B[0], c.B[3], 4, 7, 144, 192);
  dblock1(43, c.B[3], c.B[0], c.B[1], 4, 6, 192);
  dblock2(49, c.B[1], c.B[2], c.B[0], c.B[3], 4, 6, 192, 240);
  dblock1(58, c.B[3], c.B[0], c.B[1], 4, 5, 240);
  dblock2(64, c.B[1], c.B[2], c.B[0], c.B[3], 4, 5, 240, 240);

  // bottleneck 1x1 (240->3) + avgpool + softmax -> d_out [4,3] f32
  head_kernel<<<4, 256, 0, stream>>>(c.B[3], W_(73), (float*)d_out, 256, 240);
  (void)in_sizes; (void)out_size;
}
